// MultiHeadedAttention_3607772529122
// MI455X (gfx1250) — matrix operand order repaired and checked by reading, not compiled, not run on hardware
//
#include <hip/hip_runtime.h>

// ---------------- types ----------------
typedef __attribute__((ext_vector_type(16))) __bf16   v16bf;
typedef __attribute__((ext_vector_type(8)))  float    v8f;
typedef __attribute__((ext_vector_type(8)))  unsigned v8u;

// ---------------- f32 <-> split bf16 (hi + residual lo) ----------------
__device__ inline unsigned short f2bf_u(float f) {
  unsigned u = __builtin_bit_cast(unsigned, f);
  unsigned r = u + 0x7FFFu + ((u >> 16) & 1u);   // round-to-nearest-even
  return (unsigned short)(r >> 16);
}
__device__ inline float bfu2f(unsigned short b) {
  unsigned u = ((unsigned)b) << 16;
  return __builtin_bit_cast(float, u);
}
__device__ inline void split2(float f, unsigned short &h, unsigned short &l) {
  h = f2bf_u(f);
  l = f2bf_u(f - bfu2f(h));
}

// triple-product split-bf16 FMA: C += Ah*Bh + Ah*Bl + Al*Bh  (~f32 accuracy)
__device__ inline v8f wmma3(v16bf ah, v16bf al, v16bf bh, v16bf bl, v8f c) {
  c = __builtin_amdgcn_wmma_f32_16x16x32_bf16(false, ah, false, bh, (short)0, c, false, false);
  c = __builtin_amdgcn_wmma_f32_16x16x32_bf16(false, ah, false, bl, (short)0, c, false, false);
  c = __builtin_amdgcn_wmma_f32_16x16x32_bf16(false, al, false, bh, (short)0, c, false, false);
  return c;
}

// ---------------- fragment loaders: one dword load per VGPR ----------------
// A-frag (16x32, M x K): p -> &plane_dw[m*ldDW + k0/2].
// ISA layout: VGPR j<4 -> K=2j+8lh ; j>=4 -> K=16+2(j-4)+8lh (pairs) => dword
// offset (j<4 ? j : 8+(j-4)) + 4lh, pairs contiguous in row-major K.
__device__ inline v16bf frag_a(const unsigned* __restrict__ p, int lh, bool valid) {
  v8u t = {};
  if (valid) {
#pragma unroll
    for (int j = 0; j < 8; ++j)
      t[j] = p[((j < 4) ? j : (8 + (j - 4))) + 4 * lh];
  }
  return __builtin_bit_cast(v16bf, t);
}
__device__ inline v16bf frag_a_lds(const unsigned* p, int lh) {
  v8u t;
#pragma unroll
  for (int j = 0; j < 8; ++j)
    t[j] = p[((j < 4) ? j : (8 + (j - 4))) + 4 * lh];
  return __builtin_bit_cast(v16bf, t);
}
// B-frag (32x16, K x N) from K-transposed plane: p -> &plane_dw[n*ldDW + k0/2].
// VGPR j<4 -> K=2j+8lh ; j>=4 -> K=16+2(j-4)+8lh (pairs) => dword offset (j<4 ? j : 8+(j-4)) + 4lh.
__device__ inline v16bf frag_b(const unsigned* __restrict__ p, int lh) {
  v8u t;
#pragma unroll
  for (int j = 0; j < 8; ++j) t[j] = p[((j < 4) ? j : (8 + (j - 4))) + 4 * lh];
  return __builtin_bit_cast(v16bf, t);
}

// ---------------- one-time conversion kernels ----------------
__global__ __launch_bounds__(256) void split_plane(const float* __restrict__ x,
                                                   unsigned short* __restrict__ hi,
                                                   unsigned short* __restrict__ lo, int n) {
  int i = blockIdx.x * 256 + threadIdx.x;
  if (i < n) { unsigned short h, l; split2(x[i], h, l); hi[i] = h; lo[i] = l; }
}
// W[k][n] (512x512 row-major) -> K-transposed planes th[n][k], tl[n][k]
__global__ __launch_bounds__(256) void split_transpose(const float* __restrict__ W,
                                                       unsigned short* __restrict__ th,
                                                       unsigned short* __restrict__ tl) {
  int idx = blockIdx.x * 256 + threadIdx.x;       // 512*512
  int k = idx >> 9, n = idx & 511;
  unsigned short h, l; split2(W[idx], h, l);
  th[n * 512 + k] = h; tl[n * 512 + k] = l;
}

// ---------------- Kernel 1: C[M,512] = A[M,512] @ W + bias, planes in/out ----------
// 256 thr = 8 waves (4m x 2n), wave = 32x32 C block. ldDW = 256 dwords (512 bf16).
__global__ __launch_bounds__(256) void proj_gemm(
    const unsigned* __restrict__ Ah, const unsigned* __restrict__ Al,
    const unsigned* __restrict__ Bth, const unsigned* __restrict__ Btl,
    const float* __restrict__ bias,
    unsigned short* __restrict__ Ch, unsigned short* __restrict__ Cl) {
  const int lane = threadIdx.x & 31, wave = threadIdx.x >> 5;
  const int lh = lane >> 4, lm = lane & 15;
  const int m0 = blockIdx.x * 128 + (wave & 3) * 32;
  const int n0 = blockIdx.y * 64 + (wave >> 2) * 32;

  v8f c[2][2] = {};
  for (int kd = 0; kd < 256; kd += 16) {          // dword K offset (32 elems/step)
    v16bf ah[2], al[2], bh[2], bl[2];
#pragma unroll
    for (int i = 0; i < 2; ++i) {
      size_t mrow = (size_t)(m0 + i * 16 + lm) * 256 + kd;
      size_t nrow = (size_t)(n0 + i * 16 + lm) * 256 + kd;
      ah[i] = frag_a(Ah + mrow, lh, true);
      al[i] = frag_a(Al + mrow, lh, true);
      bh[i] = frag_b(Bth + nrow, lh);
      bl[i] = frag_b(Btl + nrow, lh);
    }
#pragma unroll
    for (int i = 0; i < 2; ++i)
#pragma unroll
      for (int jn = 0; jn < 2; ++jn)
        c[i][jn] = wmma3(ah[i], al[i], bh[jn], bl[jn], c[i][jn]);
  }
#pragma unroll
  for (int i = 0; i < 2; ++i)
#pragma unroll
    for (int jn = 0; jn < 2; ++jn) {
      int n = n0 + jn * 16 + lm;
      float bb = bias[n];
#pragma unroll
      for (int r = 0; r < 8; ++r) {
        size_t m = (size_t)(m0 + i * 16 + lh * 8 + r);
        unsigned short h, l; split2(c[i][jn][r] + bb, h, l);
        Ch[m * 512 + n] = h; Cl[m * 512 + n] = l;
      }
    }
}

// ---------------- Kernel 2: B_v = V_v @ Wo per head, stored K-transposed ----------
// Bpt planes: [v][n=512][k=128] bf16, k-pairs contiguous => stage-C B dword loads.
// Rows f=12..15 exact zeros via zeroed A lanes.
__global__ __launch_bounds__(256) void bprep(
    const unsigned* __restrict__ Vh, const unsigned* __restrict__ Vl,
    const unsigned* __restrict__ Wth, const unsigned* __restrict__ Wtl,
    unsigned* __restrict__ Bpth, unsigned* __restrict__ Bptl) {
  const int v = blockIdx.x;
  const int lane = threadIdx.x & 31, h = threadIdx.x >> 5;  // wave = head
  const int lh = lane >> 4, lm = lane & 15;
  const bool valid = lm < 12;
  const size_t row = (size_t)(v * 12 + (valid ? lm : 0)) * 256;

  v16bf ah[2], al[2];
#pragma unroll
  for (int ks = 0; ks < 2; ++ks) {
    ah[ks] = frag_a(Vh + row + h * 32 + ks * 16, lh, valid);
    al[ks] = frag_a(Vl + row + h * 32 + ks * 16, lh, valid);
  }
  for (int nt = 0; nt < 32; ++nt) {
    int n = nt * 16 + lm;
    v8f c = {};
#pragma unroll
    for (int ks = 0; ks < 2; ++ks) {
      v16bf bh = frag_b(Wth + (size_t)n * 256 + h * 32 + ks * 16, lh);
      v16bf bl = frag_b(Wtl + (size_t)n * 256 + h * 32 + ks * 16, lh);
      c = wmma3(ah[ks], al[ks], bh, bl, c);
    }
    size_t base = ((size_t)v * 512 + n) * 64 + h * 8 + lh * 4;  // dword idx, k row=64dw
#pragma unroll
    for (int r = 0; r < 8; r += 2) {
      unsigned short h0, l0, h1, l1;
      split2(c[r], h0, l0); split2(c[r + 1], h1, l1);
      Bpth[base + r / 2] = (unsigned)h0 | ((unsigned)h1 << 16);
      Bptl[base + r / 2] = (unsigned)l0 | ((unsigned)l1 << 16);
    }
  }
}

// ---------------- Kernel 3: fused logits -> softmax -> out = P @ B_v + bo ---------
__global__ __launch_bounds__(256) void attn_fused(
    const unsigned* __restrict__ Qh, const unsigned* __restrict__ Ql,  // q planes (row-major == B-transposed)
    const unsigned* __restrict__ Kh, const unsigned* __restrict__ Kl,
    const unsigned* __restrict__ Bpth, const unsigned* __restrict__ Bptl,
    const float* __restrict__ bo, float* __restrict__ Out) {
  __shared__ float lg[8][64][16];    // [head][t][f]        32 KB
  __shared__ unsigned Ph[64][64];    // weight hi plane dw  16 KB
  __shared__ unsigned Pl[64][64];    // weight lo plane dw  16 KB
  const int v = blockIdx.x;
  const int t0 = blockIdx.y * 64;
  const int lane = threadIdx.x & 31, wave = threadIdx.x >> 5;
  const int lh = lane >> 4, lm = lane & 15;
  const int nbase = wave * 64;

  // ---- Prefetch this wave's private B_v slice (cols nbase..nbase+63, 32 KB)
  // so it's cache-resident by the time Phase C issues its first B fragments.
  {
    const unsigned* pfb = Bpth + ((size_t)v * 512 + nbase) * 64;
    const unsigned* pfl = Bptl + ((size_t)v * 512 + nbase) * 64;
#pragma unroll
    for (int it = 0; it < 4; ++it) {               // 128B line per prefetch
      __builtin_prefetch(pfb + (it * 32 + lane) * 32, 0, 3);
      __builtin_prefetch(pfl + (it * 32 + lane) * 32, 0, 3);
    }
  }

  // ---- Phase A: logits[h][f][t] = (K_vh @ Q_h^T)/8, head h = wave
  {
    const int h = wave;
    const bool valid = lm < 12;
    const size_t row = (size_t)(v * 12 + (valid ? lm : 0)) * 256;
    v16bf ah[2], al[2];
#pragma unroll
    for (int ks = 0; ks < 2; ++ks) {
      ah[ks] = frag_a(Kh + row + h * 32 + ks * 16, lh, valid);
      al[ks] = frag_a(Kl + row + h * 32 + ks * 16, lh, valid);
    }
#pragma unroll
    for (int nt = 0; nt < 4; ++nt) {
      int t = t0 + nt * 16 + lm;
      v8f c = {};
#pragma unroll
      for (int ks = 0; ks < 2; ++ks) {
        v16bf bh = frag_b(Qh + (size_t)t * 256 + h * 32 + ks * 16, lh);
        v16bf bl = frag_b(Ql + (size_t)t * 256 + h * 32 + ks * 16, lh);
        c = wmma3(ah[ks], al[ks], bh, bl, c);
      }
#pragma unroll
      for (int r = 0; r < 8; ++r)
        lg[h][nt * 16 + lm][lh * 8 + r] = c[r] * 0.125f;   // 1/sqrt(64)
    }
  }
  __syncthreads();

  // ---- Phase B: softmax over f (12); write weight hi/lo planes, pad f>=12 with 0
  for (int col = threadIdx.x; col < 512; col += 256) {
    int h = col >> 6, t = col & 63;
    float m = -1e30f;
#pragma unroll
    for (int f = 0; f < 12; ++f) m = fmaxf(m, lg[h][t][f]);
    float e[12], s = 0.f;
#pragma unroll
    for (int f = 0; f < 12; ++f) { e[f] = __expf(lg[h][t][f] - m); s += e[f]; }
    float inv = 1.f / s;
#pragma unroll
    for (int fp = 0; fp < 6; ++fp) {
      unsigned short h0, l0, h1, l1;
      split2(e[2 * fp] * inv, h0, l0);
      split2(e[2 * fp + 1] * inv, h1, l1);
      Ph[t][h * 8 + fp] = (unsigned)h0 | ((unsigned)h1 << 16);
      Pl[t][h * 8 + fp] = (unsigned)l0 | ((unsigned)l1 << 16);
    }
    Ph[t][h * 8 + 6] = 0u; Ph[t][h * 8 + 7] = 0u;
    Pl[t][h * 8 + 6] = 0u; Pl[t][h * 8 + 7] = 0u;
  }
  __syncthreads();

  // ---- Phase C: out[64,512] = P[64,128] @ B_v[128,512] + bo; wave owns 64 n-cols
#pragma unroll
  for (int mp = 0; mp < 2; ++mp) {
    v8f c[2][4] = {};
#pragma unroll
    for (int ks = 0; ks < 4; ++ks) {
      v16bf ah[2], al[2];
#pragma unroll
      for (int i = 0; i < 2; ++i) {
        int t = (mp * 2 + i) * 16 + lm;
        ah[i] = frag_a_lds(&Ph[t][ks * 16], lh);
        al[i] = frag_a_lds(&Pl[t][ks * 16], lh);
      }
#pragma unroll
      for (int nt = 0; nt < 4; ++nt) {
        int n = nbase + nt * 16 + lm;
        size_t rb = ((size_t)v * 512 + n) * 64 + ks * 16;
        v16bf bh = frag_b(Bpth + rb, lh);
        v16bf bl = frag_b(Bptl + rb, lh);
#pragma unroll
        for (int i = 0; i < 2; ++i)
          c[i][nt] = wmma3(ah[i], al[i], bh, bl, c[i][nt]);
      }
    }
#pragma unroll
    for (int i = 0; i < 2; ++i)
#pragma unroll
      for (int nt = 0; nt < 4; ++nt) {
        int n = nbase + nt * 16 + lm;
        float bb = bo[n];
#pragma unroll
        for (int r = 0; r < 8; ++r) {
          int t = t0 + (mp * 2 + i) * 16 + lh * 8 + r;
          // streaming 536 MB result: non-temporal so it doesn't sweep L2
          __builtin_nontemporal_store(c[i][nt][r] + bb,
                                      &Out[((size_t)v * 512 + t) * 512 + n]);
        }
      }
  }
}

// ---------------- Host-side launch ----------------
extern "C" void kernel_launch(void* const* d_in, const int* in_sizes, int n_in,
                              void* d_out, int out_size, void* d_ws, size_t ws_size,
                              hipStream_t stream) {
  const float* text  = (const float*)d_in[0];
  const float* video = (const float*)d_in[1];   // [512,12,512] -> rows v*12+f
  const float* Wq = (const float*)d_in[2];
  const float* bq = (const float*)d_in[3];
  const float* Wk = (const float*)d_in[4];
  const float* bk = (const float*)d_in[5];
  const float* Wv = (const float*)d_in[6];
  const float* bv_ = (const float*)d_in[7];
  const float* Wo = (const float*)d_in[8];
  const float* bo = (const float*)d_in[9];
  float* out = (float*)d_out;

  // workspace: all bf16 planes (unsigned short), 4B-aligned sub-allocations (~178 MB)
  unsigned short* w = (unsigned short*)d_ws;
  size_t o = 0;
  auto alloc = [&](size_t n) { unsigned short* p = w + o; o += n; return p; };
  const size_t SQ = 512u * 512u;          // 262144
  const size_t SV = 6144u * 512u;         // 3145728
  unsigned short* tAh = alloc(SQ);  unsigned short* tAl = alloc(SQ);
  unsigned short* vAh = alloc(SV);  unsigned short* vAl = alloc(SV);
  unsigned short* Wqth = alloc(SQ); unsigned short* Wqtl = alloc(SQ);
  unsigned short* Wkth = alloc(SQ); unsigned short* Wktl = alloc(SQ);
  unsigned short* Wvth = alloc(SQ); unsigned short* Wvtl = alloc(SQ);
  unsigned short* Woth = alloc(SQ); unsigned short* Wotl = alloc(SQ);
  unsigned short* qh = alloc(SQ);   unsigned short* ql = alloc(SQ);
  unsigned short* kh = alloc(SV);   unsigned short* kl = alloc(SV);
  unsigned short* vh = alloc(SV);   unsigned short* vl = alloc(SV);
  unsigned* Bpth = (unsigned*)(w + o); o += 512u * 512u * 128u;   // 33554432 us
  unsigned* Bptl = (unsigned*)(w + o); o += 512u * 512u * 128u;

  dim3 blk(256);
  // one-time conversions
  split_plane<<<dim3(1024),  blk, 0, stream>>>(text,  tAh, tAl, (int)SQ);
  split_plane<<<dim3(12288), blk, 0, stream>>>(video, vAh, vAl, (int)SV);
  split_transpose<<<dim3(1024), blk, 0, stream>>>(Wq, Wqth, Wqtl);
  split_transpose<<<dim3(1024), blk, 0, stream>>>(Wk, Wkth, Wktl);
  split_transpose<<<dim3(1024), blk, 0, stream>>>(Wv, Wvth, Wvtl);
  split_transpose<<<dim3(1024), blk, 0, stream>>>(Wo, Woth, Wotl);
  // projections (outputs as hi/lo planes)
  proj_gemm<<<dim3(4, 8),  blk, 0, stream>>>((const unsigned*)tAh, (const unsigned*)tAl,
                                             (const unsigned*)Wqth, (const unsigned*)Wqtl,
                                             bq, qh, ql);
  proj_gemm<<<dim3(48, 8), blk, 0, stream>>>((const unsigned*)vAh, (const unsigned*)vAl,
                                             (const unsigned*)Wkth, (const unsigned*)Wktl,
                                             bk, kh, kl);
  proj_gemm<<<dim3(48, 8), blk, 0, stream>>>((const unsigned*)vAh, (const unsigned*)vAl,
                                             (const unsigned*)Wvth, (const unsigned*)Wvtl,
                                             bv_, vh, vl);
  // B_v = V_v @ Wo (K-transposed packed planes)
  bprep<<<dim3(512), blk, 0, stream>>>((const unsigned*)vh, (const unsigned*)vl,
                                       (const unsigned*)Woth, (const unsigned*)Wotl,
                                       Bpth, Bptl);
  // fused attention + output projection
  attn_fused<<<dim3(512, 8), blk, 0, stream>>>((const unsigned*)qh, (const unsigned*)ql,
                                               (const unsigned*)kh, (const unsigned*)kl,
                                               Bpth, Bptl, bo, out);
}
